// EnhancedTemporalGNN_30202210025474
// MI455X (gfx1250) — matrix operand order repaired and checked by reading, not compiled, not run on hardware
//
#include <hip/hip_runtime.h>
#include <hip/hip_bf16.h>
#include <math.h>

typedef __attribute__((ext_vector_type(16))) __bf16 v16bf;
typedef __attribute__((ext_vector_type(8)))  __bf16 v8bf;
typedef __attribute__((ext_vector_type(8)))  float  v8f;

#define NN 8192
#define EE 65536
#define DDIM 256
#define HH 512
#define LLAY 6
#define GHN 4
#define TCLS 20
#define ETOT (EE + NN)
#define BKST 32

__device__ inline float gelu_f(float x) { return 0.5f * x * (1.0f + erff(x * 0.70710678118654752f)); }
__device__ inline float sigmoid_f(float x) { return 1.0f / (1.0f + __expf(-x)); }

__device__ inline void atomicMaxF(float* addr, float val) {
    int* ai = (int*)addr;
    int old = __float_as_int(*addr);
    while (__int_as_float(old) < val) {
        int prev = atomicCAS(ai, old, __float_as_int(val));
        if (prev == old) break;
        old = prev;
    }
}

__device__ inline float blockReduceSum(float v) {
    __shared__ float sh[9];
    int lane = threadIdx.x & 31, w = threadIdx.x >> 5;
#pragma unroll
    for (int o = 16; o; o >>= 1) v += __shfl_xor(v, o);
    if (lane == 0) sh[w] = v;
    __syncthreads();
    if (w == 0) {
        float t = (threadIdx.x < (blockDim.x >> 5)) ? sh[threadIdx.x] : 0.f;
#pragma unroll
        for (int o = 4; o; o >>= 1) t += __shfl_xor(t, o);
        if (lane == 0) sh[8] = t;
    }
    __syncthreads();
    float r = sh[8];
    __syncthreads();
    return r;
}

// ---------------------------------------------------------------------------
// WMMA tile core v2: 128(M) x 128(N) block tile, K-steps of 32, bf16 -> f32.
// 8 waves in 4(M) x 2(N); each wave computes 32x64 = 2x4 v_wmma_f32_16x16x32_bf16
// per K-step. LDS double-buffered (one barrier per K-step); global_prefetch_b8
// issued two K-steps ahead.
// A fragment (16-bit A 16x32): m=lane&15, kg=lane>>4;
//   elems 0..7 = K[kg*8 ..], elems 8..15 = K[16+kg*8 ..].
// B fragment (16-bit B 32x16): n=lane&15, elems 0..7 = K[kg*8 ..], elems 8..15 = K[16+kg*8 ..] (same as A).
// ---------------------------------------------------------------------------
__device__ inline void ldg_tiles(const __bf16* __restrict__ A, const __bf16* __restrict__ B,
                                 int K, int bm0, int bn0, int k0,
                                 v8bf (&ra)[2], v8bf (&rb)[2]) {
    const int tid = threadIdx.x;
#pragma unroll
    for (int i = 0; i < 2; i++) {
        int c = tid + i * 256;
        int r = c >> 2, cc = (c & 3) * 8;
        ra[i] = *(const v8bf*)(A + (size_t)(bm0 + r) * K + k0 + cc);
        rb[i] = *(const v8bf*)(B + (size_t)(bn0 + r) * K + k0 + cc);
    }
}

__device__ inline void stg_tiles(__bf16 (&As)[128][40], __bf16 (&Bs)[128][40],
                                 v8bf (&ra)[2], v8bf (&rb)[2]) {
    const int tid = threadIdx.x;
#pragma unroll
    for (int i = 0; i < 2; i++) {
        int c = tid + i * 256;
        int r = c >> 2, cc = (c & 3) * 8;
        *(v8bf*)&As[r][cc] = ra[i];
        *(v8bf*)&Bs[r][cc] = rb[i];
    }
}

__device__ inline void compute_step(__bf16 (&As)[128][40], __bf16 (&Bs)[128][40],
                                    v8f (&acc)[2][4]) {
    const int lane = threadIdx.x & 31;
    const int wave = threadIdx.x >> 5;
    const int wm = wave >> 1, wn = wave & 1;
    const int mrow = lane & 15, kg = lane >> 4;
    v16bf af[2], bfr[4];
#pragma unroll
    for (int i = 0; i < 2; i++) {
        const __bf16* ap = &As[wm * 32 + i * 16 + mrow][0];
        union { v16bf v; v8bf h[2]; } u;
        u.h[0] = *(const v8bf*)(ap + kg * 8);
        u.h[1] = *(const v8bf*)(ap + 16 + kg * 8);
        af[i] = u.v;
    }
#pragma unroll
    for (int j = 0; j < 4; j++) {
        const __bf16* bp = &Bs[wn * 64 + j * 16 + mrow][0];
        union { v16bf v; v8bf h[2]; } u;
        u.h[0] = *(const v8bf*)(bp + kg * 8);
        u.h[1] = *(const v8bf*)(bp + 16 + kg * 8);
        bfr[j] = u.v;
    }
#pragma unroll
    for (int i = 0; i < 2; i++)
#pragma unroll
        for (int j = 0; j < 4; j++)
            acc[i][j] = __builtin_amdgcn_wmma_f32_16x16x32_bf16(
                false, af[i], false, bfr[j], (short)0, acc[i][j], false, false);
}

__device__ inline void wmma_tile_core(const __bf16* __restrict__ A, const __bf16* __restrict__ B,
                                      int K, int bm0, int bn0,
                                      __bf16 (&As)[2][128][40], __bf16 (&Bs)[2][128][40],
                                      v8f (&acc)[2][4]) {
    v8bf ra[2], rb[2];
    ldg_tiles(A, B, K, bm0, bn0, 0, ra, rb);
    stg_tiles(As[0], Bs[0], ra, rb);
    __syncthreads();
    int cur = 0;
    for (int k0 = BKST; k0 < K; k0 += BKST) {
        ldg_tiles(A, B, K, bm0, bn0, k0, ra, rb);
        if (k0 + BKST < K) {
            const int tid = threadIdx.x;
            int r = tid >> 2, cc = (tid & 3) * 8;
            __builtin_prefetch(A + (size_t)(bm0 + r) * K + k0 + BKST + cc, 0, 1);
            __builtin_prefetch(B + (size_t)(bn0 + r) * K + k0 + BKST + cc, 0, 1);
        }
        compute_step(As[cur], Bs[cur], acc);
        stg_tiles(As[cur ^ 1], Bs[cur ^ 1], ra, rb);
        __syncthreads();
        cur ^= 1;
    }
    compute_step(As[cur], Bs[cur], acc);
}

// C[M,Nout] = act( A[M,K](bf16) @ B[Nout,K]^T(bf16) + bias + resid ), f32/bf16 out
__global__ __launch_bounds__(256) void wmma_gemm_kernel(
    const __bf16* __restrict__ A, const __bf16* __restrict__ B,
    const float* __restrict__ bias, const float* __restrict__ resid,
    float* __restrict__ Cf, __bf16* __restrict__ Cb,
    int M, int Nout, int K, int act) {
    __shared__ __bf16 As[2][128][40];
    __shared__ __bf16 Bs[2][128][40];
    v8f acc[2][4];
    v8f z = {0.f, 0.f, 0.f, 0.f, 0.f, 0.f, 0.f, 0.f};
#pragma unroll
    for (int i = 0; i < 2; i++)
#pragma unroll
        for (int j = 0; j < 4; j++) acc[i][j] = z;
    const int bm0 = blockIdx.y * 128;
    const int bn0 = blockIdx.x * 128;
    wmma_tile_core(A, B, K, bm0, bn0, As, Bs, acc);

    const int lane = threadIdx.x & 31;
    const int wave = threadIdx.x >> 5;
    const int wm = wave >> 1, wn = wave & 1;
    const int rowhalf = (lane >> 4) * 8;
    const int coln = lane & 15;
#pragma unroll
    for (int i = 0; i < 2; i++)
#pragma unroll
        for (int j = 0; j < 4; j++) {
            int n = bn0 + wn * 64 + j * 16 + coln;
            float bv = bias ? bias[n] : 0.0f;
#pragma unroll
            for (int r = 0; r < 8; r++) {
                int m = bm0 + wm * 32 + i * 16 + rowhalf + r;
                float v = acc[i][j][r] + bv;
                if (resid) v += resid[(size_t)m * Nout + n];
                if (act) v = gelu_f(v);
                if (Cf) Cf[(size_t)m * Nout + n] = v;
                if (Cb) Cb[(size_t)m * Nout + n] = (__bf16)v;
            }
        }
}

// Streaming S = X @ X^T passes: 0=rowmax, 1=rowsumexp, 2=prob colsum
__global__ __launch_bounds__(256) void score_pass_kernel(
    const __bf16* __restrict__ X, int Nn, int K,
    float* __restrict__ rowmax, float* __restrict__ rowsum,
    float* __restrict__ colsum, int pass) {
    __shared__ __bf16 As[2][128][40];
    __shared__ __bf16 Bs[2][128][40];
    v8f acc[2][4];
    v8f z = {0.f, 0.f, 0.f, 0.f, 0.f, 0.f, 0.f, 0.f};
#pragma unroll
    for (int i = 0; i < 2; i++)
#pragma unroll
        for (int j = 0; j < 4; j++) acc[i][j] = z;
    const int bm0 = blockIdx.y * 128;
    const int bn0 = blockIdx.x * 128;
    wmma_tile_core(X, X, K, bm0, bn0, As, Bs, acc);

    const int lane = threadIdx.x & 31;
    const int wave = threadIdx.x >> 5;
    const int wm = wave >> 1, wn = wave & 1;
    const int rowhalf = (lane >> 4) * 8;
    const int coln = lane & 15;

    if (pass == 0) {
#pragma unroll
        for (int i = 0; i < 2; i++)
#pragma unroll
            for (int r = 0; r < 8; r++) {
                int m = bm0 + wm * 32 + i * 16 + rowhalf + r;
                float v = fmaxf(fmaxf(acc[i][0][r], acc[i][1][r]),
                                fmaxf(acc[i][2][r], acc[i][3][r]));
#pragma unroll
                for (int o = 1; o < 16; o <<= 1) v = fmaxf(v, __shfl_xor(v, o));
                if (coln == 0) atomicMaxF(&rowmax[m], v);
            }
    } else if (pass == 1) {
#pragma unroll
        for (int i = 0; i < 2; i++)
#pragma unroll
            for (int r = 0; r < 8; r++) {
                int m = bm0 + wm * 32 + i * 16 + rowhalf + r;
                float rm = rowmax[m];
                float v = __expf(acc[i][0][r] - rm) + __expf(acc[i][1][r] - rm) +
                          __expf(acc[i][2][r] - rm) + __expf(acc[i][3][r] - rm);
#pragma unroll
                for (int o = 1; o < 16; o <<= 1) v += __shfl_xor(v, o);
                if (coln == 0) atomicAdd(&rowsum[m], v);
            }
    } else {
        float cs[4] = {0.f, 0.f, 0.f, 0.f};
#pragma unroll
        for (int i = 0; i < 2; i++)
#pragma unroll
            for (int r = 0; r < 8; r++) {
                int m = bm0 + wm * 32 + i * 16 + rowhalf + r;
                float rm = rowmax[m];
                float inv = 1.0f / rowsum[m];
#pragma unroll
                for (int j = 0; j < 4; j++) cs[j] += __expf(acc[i][j][r] - rm) * inv;
            }
#pragma unroll
        for (int j = 0; j < 4; j++) cs[j] += __shfl_xor(cs[j], 16);
        if (lane < 16) {
#pragma unroll
            for (int j = 0; j < 4; j++)
                atomicAdd(&colsum[bn0 + wn * 64 + j * 16 + coln], cs[j]);
        }
    }
}

// ----------------------------- elementwise/helpers --------------------------
__global__ void cvt_bf16_kernel(const float* __restrict__ s, __bf16* __restrict__ d, size_t n) {
    size_t i = (size_t)blockIdx.x * blockDim.x + threadIdx.x;
    size_t stride = (size_t)gridDim.x * blockDim.x;
    for (; i < n; i += stride) d[i] = (__bf16)s[i];
}

__global__ void fill_kernel(float* __restrict__ p, float v, size_t n) {
    size_t i = (size_t)blockIdx.x * blockDim.x + threadIdx.x;
    if (i < n) p[i] = v;
}

__global__ __launch_bounds__(256) void layernorm_gelu_kernel(
    const float* __restrict__ X, const float* __restrict__ g, const float* __restrict__ b,
    float* __restrict__ Of, __bf16* __restrict__ Ob, int Hh) {
    int row = blockIdx.x;
    const float* x = X + (size_t)row * Hh;
    int t = threadIdx.x;
    float v0 = x[t], v1 = x[t + 256];
    float s = blockReduceSum(v0 + v1);
    float s2 = blockReduceSum(v0 * v0 + v1 * v1);
    float mu = s / Hh;
    float var = s2 / Hh - mu * mu;
    float rstd = rsqrtf(var + 1e-5f);
    float y0 = gelu_f((v0 - mu) * rstd * g[t] + b[t]);
    float y1 = gelu_f((v1 - mu) * rstd * g[t + 256] + b[t + 256]);
    Of[(size_t)row * Hh + t] = y0;
    Of[(size_t)row * Hh + t + 256] = y1;
    Ob[(size_t)row * Hh + t] = (__bf16)y0;
    Ob[(size_t)row * Hh + t + 256] = (__bf16)y1;
}

__global__ void time1_kernel(const float* __restrict__ ts, const float* __restrict__ W1,
                             const float* __restrict__ b1, __bf16* __restrict__ out, int N, int Hh) {
    int i = blockIdx.x * blockDim.x + threadIdx.x;
    if (i >= N * Hh) return;
    int n = i / Hh, h = i % Hh;
    out[i] = (__bf16)gelu_f(ts[n] * W1[h] + b1[h]);
}

__global__ void deg_edge_kernel(const int* __restrict__ ei, float* __restrict__ deg, int E) {
    int e = blockIdx.x * blockDim.x + threadIdx.x;
    if (e < E) atomicAdd(&deg[ei[E + e]], 1.0f);
}

__global__ void rsqrt_kernel(const float* __restrict__ deg, float* __restrict__ dis, int n) {
    int i = blockIdx.x * blockDim.x + threadIdx.x;
    if (i < n) { float d = deg[i]; dis[i] = d > 0.f ? rsqrtf(d) : 0.f; }
}

__global__ void gcn_scatter_kernel(const int* __restrict__ ei, const float* __restrict__ dis,
                                   const float* __restrict__ xw, float* __restrict__ out,
                                   int E, int N, int Hh) {
    int idx = blockIdx.x * blockDim.x + threadIdx.x;
    int per = Hh / 4;
    int e = idx / per;
    if (e >= E + N) return;
    int f = (idx % per) * 4;
    int r, c;
    if (e < E) { r = ei[e]; c = ei[E + e]; } else { r = e - E; c = r; }
    float w = dis[r] * dis[c];
    const float4 v = *(const float4*)(xw + (size_t)r * Hh + f);
    float* o = out + (size_t)c * Hh + f;
    atomicAdd(o + 0, w * v.x); atomicAdd(o + 1, w * v.y);
    atomicAdd(o + 2, w * v.z); atomicAdd(o + 3, w * v.w);
}

__global__ void bias_gelu_kernel(float* __restrict__ x, const float* __restrict__ b,
                                 size_t total, int Hh) {
    size_t i = (size_t)blockIdx.x * blockDim.x + threadIdx.x;
    if (i >= total) return;
    x[i] = gelu_f(x[i] + b[i % Hh]);
}

__global__ __launch_bounds__(256) void gat_scores_kernel(
    const float* __restrict__ xl, const float* __restrict__ as_, const float* __restrict__ ad_,
    float* __restrict__ asum, float* __restrict__ adsum,
    float* __restrict__ segmax, float* __restrict__ segsum, int N, int Hh, int GHh) {
    int gwave = (blockIdx.x * blockDim.x + threadIdx.x) >> 5;
    int lane = threadIdx.x & 31;
    if (gwave >= N * GHh) return;
    int n = gwave / GHh, h = gwave % GHh;
    const float* xr = xl + (size_t)n * (GHh * Hh) + h * Hh;
    const float* av = as_ + h * Hh;
    const float* dv = ad_ + h * Hh;
    float s = 0.f, d = 0.f;
    for (int k = lane; k < Hh; k += 32) { float x = xr[k]; s += x * av[k]; d += x * dv[k]; }
#pragma unroll
    for (int o = 16; o; o >>= 1) { s += __shfl_xor(s, o); d += __shfl_xor(d, o); }
    if (lane == 0) {
        asum[n * GHh + h] = s; adsum[n * GHh + h] = d;
        segmax[n * GHh + h] = -1e30f; segsum[n * GHh + h] = 0.f;
    }
}

__global__ void gat_max_kernel(const int* __restrict__ ei, const float* __restrict__ asum,
                               const float* __restrict__ adsum, float* __restrict__ segmax,
                               int E, int N) {
    int e = blockIdx.x * blockDim.x + threadIdx.x;
    if (e >= E + N) return;
    int r, c;
    if (e < E) { r = ei[e]; c = ei[E + e]; } else { r = e - E; c = r; }
#pragma unroll
    for (int h = 0; h < GHN; h++) {
        float a = asum[r * GHN + h] + adsum[c * GHN + h];
        a = a > 0.f ? a : 0.2f * a;
        atomicMaxF(&segmax[c * GHN + h], a);
    }
}

__global__ void gat_sum_kernel(const int* __restrict__ ei, const float* __restrict__ asum,
                               const float* __restrict__ adsum, const float* __restrict__ segmax,
                               float* __restrict__ segsum, int E, int N) {
    int e = blockIdx.x * blockDim.x + threadIdx.x;
    if (e >= E + N) return;
    int r, c;
    if (e < E) { r = ei[e]; c = ei[E + e]; } else { r = e - E; c = r; }
#pragma unroll
    for (int h = 0; h < GHN; h++) {
        float a = asum[r * GHN + h] + adsum[c * GHN + h];
        a = a > 0.f ? a : 0.2f * a;
        atomicAdd(&segsum[c * GHN + h], __expf(a - segmax[c * GHN + h]));
    }
}

__global__ void gat_alpha_kernel(const int* __restrict__ ei, const float* __restrict__ asum,
                                 const float* __restrict__ adsum, const float* __restrict__ segmax,
                                 const float* __restrict__ segsum, float* __restrict__ ealpha,
                                 int E, int N) {
    int e = blockIdx.x * blockDim.x + threadIdx.x;
    if (e >= E + N) return;
    int r, c;
    if (e < E) { r = ei[e]; c = ei[E + e]; } else { r = e - E; c = r; }
#pragma unroll
    for (int h = 0; h < GHN; h++) {
        float a = asum[r * GHN + h] + adsum[c * GHN + h];
        a = a > 0.f ? a : 0.2f * a;
        ealpha[(size_t)e * GHN + h] = __expf(a - segmax[c * GHN + h]) / (segsum[c * GHN + h] + 1e-16f);
    }
}

__global__ void gat_agg_kernel(const int* __restrict__ ei, const float* __restrict__ ealpha,
                               const float* __restrict__ xl, float* __restrict__ out,
                               int E, int N, int Hh) {
    int idx = blockIdx.x * blockDim.x + threadIdx.x;
    int per = Hh / 4;
    int e = idx / per;
    if (e >= E + N) return;
    int f = (idx % per) * 4;
    int r, c;
    if (e < E) { r = ei[e]; c = ei[E + e]; } else { r = e - E; c = r; }
    float ax = 0.f, ay = 0.f, az = 0.f, aw = 0.f;
#pragma unroll
    for (int h = 0; h < GHN; h++) {
        float al = ealpha[(size_t)e * GHN + h] * 0.25f;
        const float4 v = *(const float4*)(xl + (size_t)r * (GHN * Hh) + h * Hh + f);
        ax += al * v.x; ay += al * v.y; az += al * v.z; aw += al * v.w;
    }
    float* o = out + (size_t)c * Hh + f;
    atomicAdd(o + 0, ax); atomicAdd(o + 1, ay); atomicAdd(o + 2, az); atomicAdd(o + 3, aw);
}

__global__ void add3_kernel(float* __restrict__ x, const float* __restrict__ a,
                            const float* __restrict__ b, __bf16* __restrict__ xb, size_t n) {
    size_t i = (size_t)blockIdx.x * blockDim.x + threadIdx.x;
    if (i >= n) return;
    float v = x[i] + a[i] + b[i];
    x[i] = v; xb[i] = (__bf16)v;
}

__global__ void concat_kernel(const float* __restrict__ x, const float* __restrict__ xt,
                              float* __restrict__ xcf, __bf16* __restrict__ xcb, int N, int Hh) {
    size_t i = (size_t)blockIdx.x * blockDim.x + threadIdx.x;
    size_t tot = (size_t)N * 2 * Hh;
    if (i >= tot) return;
    int n = (int)(i / (2 * Hh));
    int h = (int)(i % (2 * Hh));
    float v = (h < Hh) ? x[(size_t)n * Hh + h] : xt[(size_t)n * Hh + (h - Hh)];
    xcf[i] = v; xcb[i] = (__bf16)v;
}

__global__ __launch_bounds__(256) void pool_gemv_kernel(
    const float* __restrict__ csum, const float* __restrict__ xcf,
    float* __restrict__ xp, int N, int C) {
    int h = blockIdx.x;
    float s = 0.f;
    for (int j = threadIdx.x; j < N; j += blockDim.x) s += csum[j] * xcf[(size_t)j * C + h];
    s = blockReduceSum(s);
    if (threadIdx.x == 0) xp[h] = s / (float)N;
}

__global__ __launch_bounds__(256) void gemv_kernel(
    const float* __restrict__ in, const float* __restrict__ W,
    const float* __restrict__ b, float* __restrict__ out, int K, int act) {
    int o = blockIdx.x;
    const float* w = W + (size_t)o * K;
    float s = 0.f;
    for (int k = threadIdx.x; k < K; k += blockDim.x) s += in[k] * w[k];
    s = blockReduceSum(s);
    if (threadIdx.x == 0) {
        float v = s + b[o];
        out[o] = act ? gelu_f(v) : v;
    }
}

__global__ void final_kernel(const float* __restrict__ thr, const float* __restrict__ risk,
                             const float* __restrict__ zd, const float* __restrict__ nov,
                             float* __restrict__ out) {
    if (threadIdx.x == 0) {
        float mx = -1e30f;
        for (int i = 0; i < TCLS; i++) mx = fmaxf(mx, thr[i]);
        float e[TCLS]; float s = 0.f;
        for (int i = 0; i < TCLS; i++) { e[i] = __expf(thr[i] - mx); s += e[i]; }
        for (int i = 0; i < TCLS; i++) out[i] = e[i] / s;
        out[TCLS + 0] = sigmoid_f(risk[0]);
        out[TCLS + 1] = sigmoid_f(zd[0]);
        out[TCLS + 2] = sigmoid_f(nov[0]);
    }
}

// ---------------------------------------------------------------------------
extern "C" void kernel_launch(void* const* d_in, const int* in_sizes, int n_in,
                              void* d_out, int out_size, void* d_ws, size_t ws_size,
                              hipStream_t stream) {
    (void)in_sizes; (void)n_in; (void)out_size; (void)ws_size;
    const float* nodef = (const float*)d_in[0];
    const int* eidx = (const int*)d_in[1];
    const float* ts = (const float*)d_in[2];
#define PF(i) ((const float*)d_in[(i)])
    const float *p_feW1 = PF(3), *p_feb1 = PF(4), *p_feg1 = PF(5), *p_febe1 = PF(6);
    const float *p_feW2 = PF(7), *p_feb2 = PF(8), *p_feg2 = PF(9), *p_febe2 = PF(10);
    const float *p_teW1 = PF(11), *p_teb1 = PF(12), *p_teW2 = PF(13), *p_teb2 = PF(14);
    const float *p_aWv = PF(15), *p_abv = PF(16), *p_aWo = PF(17), *p_abo = PF(18);
    const float *p_gcnW = PF(19), *p_gcnb = PF(20);
    const float *p_gatW = PF(21), *p_gatas = PF(22), *p_gatad = PF(23), *p_gatb = PF(24);
    const float *p_tcWv = PF(25), *p_tcbv = PF(26), *p_tcWo = PF(27), *p_tcbo = PF(28);
    const float *p_thrW1 = PF(29), *p_thrb1 = PF(30), *p_thrW2 = PF(31), *p_thrb2 = PF(32),
                *p_thrW3 = PF(33), *p_thrb3 = PF(34);
    const float *p_rW1 = PF(35), *p_rb1 = PF(36), *p_rW2 = PF(37), *p_rb2 = PF(38),
                *p_rW3 = PF(39), *p_rb3 = PF(40);
    const float *p_zW1 = PF(41), *p_zb1 = PF(42), *p_zW2 = PF(43), *p_zb2 = PF(44),
                *p_zW3 = PF(45), *p_zb3 = PF(46);
    const float *p_nW1 = PF(47), *p_nb1 = PF(48), *p_nW2 = PF(49), *p_nb2 = PF(50);

    // workspace bump allocator
    size_t cur = 0;
    auto take = [&](size_t elems, size_t esz) -> void* {
        void* r = (char*)d_ws + cur;
        cur += (elems * esz + 255) & ~(size_t)255;
        return r;
    };
    __bf16* nfbf = (__bf16*)take((size_t)NN * DDIM, 2);
    __bf16* w_fe1 = (__bf16*)take((size_t)HH * DDIM, 2);
    __bf16* w_fe2 = (__bf16*)take((size_t)HH * HH, 2);
    __bf16* w_te2 = (__bf16*)take((size_t)HH * HH, 2);
    __bf16* w_awv = (__bf16*)take((size_t)LLAY * HH * HH, 2);
    __bf16* w_awo = (__bf16*)take((size_t)LLAY * HH * HH, 2);
    __bf16* w_gcn = (__bf16*)take((size_t)LLAY * HH * HH, 2);
    __bf16* w_gat = (__bf16*)take((size_t)LLAY * GHN * HH * HH, 2);
    __bf16* w_tcv = (__bf16*)take((size_t)HH * HH, 2);
    __bf16* w_tco = (__bf16*)take((size_t)HH * HH, 2);
    float* xf = (float*)take((size_t)NN * HH, 4);
    __bf16* xbf = (__bf16*)take((size_t)NN * HH, 2);
    float* yf = (float*)take((size_t)NN * HH, 4);
    __bf16* ybf = (__bf16*)take((size_t)NN * HH, 2);
    __bf16* tbf = (__bf16*)take((size_t)NN * HH, 2);
    float* gcnf = (float*)take((size_t)NN * HH, 4);
    float* gatf = (float*)take((size_t)NN * HH, 4);
    float* xlf = (float*)take((size_t)NN * GHN * HH, 4);
    float* deg = (float*)take(NN, 4);
    float* dis = (float*)take(NN, 4);
    float* asum = (float*)take((size_t)NN * GHN, 4);
    float* adsum = (float*)take((size_t)NN * GHN, 4);
    float* segmax = (float*)take((size_t)NN * GHN, 4);
    float* segsum = (float*)take((size_t)NN * GHN, 4);
    float* ealpha = (float*)take((size_t)ETOT * GHN, 4);
    float* xcf = (float*)take((size_t)NN * 2 * HH, 4);
    __bf16* xcbf = (__bf16*)take((size_t)NN * 2 * HH, 2);
    float* rowmax = (float*)take(NN, 4);
    float* rowsum = (float*)take(NN, 4);
    float* colsum = (float*)take(NN, 4);
    float* xp = (float*)take(2 * HH, 4);
    float* h1 = (float*)take(HH, 4);
    float* h2 = (float*)take(HH / 2, 4);
    float* lthr = (float*)take(TCLS, 4);
    float* lrisk = (float*)take(1, 4);
    float* lzd = (float*)take(1, 4);
    float* lnov = (float*)take(1, 4);

    auto cvt = [&](const float* s, __bf16* dst, size_t n) {
        int blocks = (int)((n + 1023) / 1024);
        cvt_bf16_kernel<<<blocks, 256, 0, stream>>>(s, dst, n);
    };
    auto fill = [&](float* p, float v, size_t n) {
        fill_kernel<<<(int)((n + 255) / 256), 256, 0, stream>>>(p, v, n);
    };
    auto gemm = [&](const __bf16* A, const __bf16* B, const float* bias, const float* resid,
                    float* Cf, __bf16* Cb, int M, int Nout, int K, int act) {
        dim3 g(Nout / 128, M / 128);
        wmma_gemm_kernel<<<g, 256, 0, stream>>>(A, B, bias, resid, Cf, Cb, M, Nout, K, act);
    };
    const size_t NH = (size_t)NN * HH;
    const int nhBlocks = (int)((NH + 255) / 256);

    // weight / input conversion to bf16
    cvt(nodef, nfbf, (size_t)NN * DDIM);
    cvt(p_feW1, w_fe1, (size_t)HH * DDIM);
    cvt(p_feW2, w_fe2, (size_t)HH * HH);
    cvt(p_teW2, w_te2, (size_t)HH * HH);
    cvt(p_aWv, w_awv, (size_t)LLAY * HH * HH);
    cvt(p_aWo, w_awo, (size_t)LLAY * HH * HH);
    cvt(p_gcnW, w_gcn, (size_t)LLAY * HH * HH);
    cvt(p_gatW, w_gat, (size_t)LLAY * GHN * HH * HH);
    cvt(p_tcWv, w_tcv, (size_t)HH * HH);
    cvt(p_tcWo, w_tco, (size_t)HH * HH);

    // feature encoder
    gemm(nfbf, w_fe1, p_feb1, nullptr, yf, nullptr, NN, HH, DDIM, 0);
    layernorm_gelu_kernel<<<NN, 256, 0, stream>>>(yf, p_feg1, p_febe1, xf, xbf, HH);
    gemm(xbf, w_fe2, p_feb2, nullptr, yf, nullptr, NN, HH, HH, 0);
    layernorm_gelu_kernel<<<NN, 256, 0, stream>>>(yf, p_feg2, p_febe2, xf, xbf, HH);

    // time encoder + residual add
    time1_kernel<<<(NN * HH) / 256, 256, 0, stream>>>(ts, p_teW1, p_teb1, tbf, NN, HH);
    gemm(tbf, w_te2, p_teb2, xf, xf, xbf, NN, HH, HH, 0);

    // 6 MHA layers (seq_len=1 -> out_proj(v_proj(x)) residual)
    for (int l = 0; l < LLAY; l++) {
        gemm(xbf, w_awv + (size_t)l * HH * HH, p_abv + l * HH, nullptr, nullptr, ybf, NN, HH, HH, 0);
        gemm(ybf, w_awo + (size_t)l * HH * HH, p_abo + l * HH, xf, xf, xbf, NN, HH, HH, 0);
    }

    // GCN degree normalization
    fill(deg, 1.0f, NN);
    deg_edge_kernel<<<(EE + 255) / 256, 256, 0, stream>>>(eidx, deg, EE);
    rsqrt_kernel<<<(NN + 255) / 256, 256, 0, stream>>>(deg, dis, NN);

    // GNN layers
    for (int l = 0; l < LLAY; l++) {
        gemm(xbf, w_gcn + (size_t)l * HH * HH, nullptr, nullptr, yf, nullptr, NN, HH, HH, 0);
        fill(gcnf, 0.f, NH);
        gcn_scatter_kernel<<<(ETOT * (HH / 4) + 255) / 256, 256, 0, stream>>>(eidx, dis, yf, gcnf, EE, NN, HH);
        bias_gelu_kernel<<<nhBlocks, 256, 0, stream>>>(gcnf, p_gcnb + l * HH, NH, HH);

        gemm(xbf, w_gat + (size_t)l * GHN * HH * HH, nullptr, nullptr, xlf, nullptr, NN, GHN * HH, HH, 0);
        gat_scores_kernel<<<(NN * GHN) / 8, 256, 0, stream>>>(
            xlf, p_gatas + (size_t)l * GHN * HH, p_gatad + (size_t)l * GHN * HH,
            asum, adsum, segmax, segsum, NN, HH, GHN);
        gat_max_kernel<<<(ETOT + 255) / 256, 256, 0, stream>>>(eidx, asum, adsum, segmax, EE, NN);
        gat_sum_kernel<<<(ETOT + 255) / 256, 256, 0, stream>>>(eidx, asum, adsum, segmax, segsum, EE, NN);
        gat_alpha_kernel<<<(ETOT + 255) / 256, 256, 0, stream>>>(eidx, asum, adsum, segmax, segsum, ealpha, EE, NN);
        fill(gatf, 0.f, NH);
        gat_agg_kernel<<<(ETOT * (HH / 4) + 255) / 256, 256, 0, stream>>>(eidx, ealpha, xlf, gatf, EE, NN, HH);
        bias_gelu_kernel<<<nhBlocks, 256, 0, stream>>>(gatf, p_gatb + l * HH, NH, HH);

        add3_kernel<<<nhBlocks, 256, 0, stream>>>(xf, gcnf, gatf, xbf, NH);
    }

    // temporal cross attention (seq_len=1)
    gemm(xbf, w_tcv, p_tcbv, nullptr, nullptr, ybf, NN, HH, HH, 0);
    gemm(ybf, w_tco, p_tcbo, nullptr, yf, nullptr, NN, HH, HH, 0);
    concat_kernel<<<(int)(((size_t)NN * 2 * HH + 255) / 256), 256, 0, stream>>>(xf, yf, xcf, xcbf, NN, HH);

    // pooled softmax attention: xp = (colsum(softmax(xc@xc^T))/N) @ xc
    fill(rowmax, -1e30f, NN);
    fill(rowsum, 0.f, NN);
    fill(colsum, 0.f, NN);
    dim3 sg(NN / 128, NN / 128);
    score_pass_kernel<<<sg, 256, 0, stream>>>(xcbf, NN, 2 * HH, rowmax, rowsum, colsum, 0);
    score_pass_kernel<<<sg, 256, 0, stream>>>(xcbf, NN, 2 * HH, rowmax, rowsum, colsum, 1);
    score_pass_kernel<<<sg, 256, 0, stream>>>(xcbf, NN, 2 * HH, rowmax, rowsum, colsum, 2);
    pool_gemv_kernel<<<2 * HH, 256, 0, stream>>>(colsum, xcf, xp, NN, 2 * HH);

    // heads
    gemv_kernel<<<HH, 256, 0, stream>>>(xp, p_thrW1, p_thrb1, h1, 2 * HH, 1);
    gemv_kernel<<<HH / 2, 256, 0, stream>>>(h1, p_thrW2, p_thrb2, h2, HH, 1);
    gemv_kernel<<<TCLS, 256, 0, stream>>>(h2, p_thrW3, p_thrb3, lthr, HH / 2, 0);

    gemv_kernel<<<HH, 256, 0, stream>>>(xp, p_rW1, p_rb1, h1, 2 * HH, 1);
    gemv_kernel<<<HH / 2, 256, 0, stream>>>(h1, p_rW2, p_rb2, h2, HH, 1);
    gemv_kernel<<<1, 256, 0, stream>>>(h2, p_rW3, p_rb3, lrisk, HH / 2, 0);

    gemv_kernel<<<HH, 256, 0, stream>>>(xp, p_zW1, p_zb1, h1, 2 * HH, 1);
    gemv_kernel<<<HH / 2, 256, 0, stream>>>(h1, p_zW2, p_zb2, h2, HH, 1);
    gemv_kernel<<<1, 256, 0, stream>>>(h2, p_zW3, p_zb3, lzd, HH / 2, 0);

    gemv_kernel<<<HH, 256, 0, stream>>>(xp, p_nW1, p_nb1, h1, 2 * HH, 1);
    gemv_kernel<<<1, 256, 0, stream>>>(h1, p_nW2, p_nb2, lnov, HH, 0);

    final_kernel<<<1, 32, 0, stream>>>(lthr, lrisk, lzd, lnov, (float*)d_out);
}
